// FrameTransformerEncoder_70849780515276
// MI455X (gfx1250) — matrix operand order repaired and checked by reading, not compiled, not run on hardware
//
#include <hip/hip_runtime.h>
#include <hip/hip_bf16.h>

// ---------------------------------------------------------------------------
// MI455X (gfx1250) implementation.
// All matmuls (4x mclin batched GEMM, QK^T, PV) run on v_wmma_f32_16x16x32_f16
// with f32 accumulation, 2x2 register blocking per wave (32x32 output block,
// 2 b128 loads per WMMA). Operands staged once in f16 in fragment-friendly
// layouts. Convs / norms / rope / softmax are f32 VALU kernels.
// ---------------------------------------------------------------------------

typedef _Float16 h16;
typedef __attribute__((ext_vector_type(16))) _Float16 v16h;
typedef __attribute__((ext_vector_type(8)))  _Float16 v8h;
typedef __attribute__((ext_vector_type(8)))  float    v8f;

#define FF 1024
#define WW 512
#define HD 128
#define NHEAD 8
#define NB 2
#define NC 8
#define NM 8

static __device__ __forceinline__ v16h ldfragA(const h16* p) {
  v8h lo = *(const v8h*)(p);
  v8h hi = *(const v8h*)(p + 8);
  return __builtin_shufflevector(lo, hi, 0, 1, 2, 3, 4, 5, 6, 7,
                                 8, 9, 10, 11, 12, 13, 14, 15);
}

#define WMMA(ACC, A, B) \
  ACC = __builtin_amdgcn_wmma_f32_16x16x32_f16(false, A, false, B, (short)0, ACC, false, false)

// ---------------------------------------------------------------------------
// f32 -> f16 elementwise convert (weights)
// ---------------------------------------------------------------------------
__global__ __launch_bounds__(256) void cvt_f32_f16(const float* __restrict__ in,
                                                   h16* __restrict__ out, int n) {
  int i = blockIdx.x * 256 + threadIdx.x;
  if (i < n) out[i] = (h16)in[i];
}

// ---------------------------------------------------------------------------
// Generic 3x3 SAME conv, NCHW / OIHW.  Optional fused exact GELU.
// Optional f32 output (b,co,f,w) and/or f16 transposed output (b,co,w,f)
// (the latter is the WMMA B-operand layout for the mclin GEMMs).
// ---------------------------------------------------------------------------
__global__ __launch_bounds__(256) void conv3x3(const float* __restrict__ in,
                                               const float* __restrict__ wgt,
                                               const float* __restrict__ bias,
                                               int Cin, int Cout,
                                               float* __restrict__ out_f,
                                               h16* __restrict__ out_t,
                                               int fuse_gelu) {
  size_t idx = (size_t)blockIdx.x * 256 + threadIdx.x;
  size_t total = (size_t)NB * Cout * FF * WW;
  if (idx >= total) return;
  int w  = (int)(idx % WW);
  int f  = (int)((idx / WW) % FF);
  int co = (int)((idx / ((size_t)WW * FF)) % Cout);
  int b  = (int)(idx / ((size_t)WW * FF * Cout));

  float acc = bias[co];
  for (int ci = 0; ci < Cin; ++ci) {
    const float* ip = in + ((size_t)(b * Cin + ci) * FF) * WW;
    const float* wp = wgt + (size_t)(co * Cin + ci) * 9;
#pragma unroll
    for (int df = -1; df <= 1; ++df) {
      int ff = f + df;
      if (ff < 0 || ff >= FF) continue;
#pragma unroll
      for (int dw = -1; dw <= 1; ++dw) {
        int ww2 = w + dw;
        if (ww2 < 0 || ww2 >= WW) continue;
        acc += ip[(size_t)ff * WW + ww2] * wp[(df + 1) * 3 + (dw + 1)];
      }
    }
  }
  if (fuse_gelu) acc = 0.5f * acc * (1.0f + erff(acc * 0.70710678118654752f));
  if (out_f) out_f[idx] = acc;
  if (out_t) out_t[((size_t)(b * Cout + co) * WW + w) * FF + f] = (h16)acc;
}

// ---------------------------------------------------------------------------
// mclin WMMA GEMM (2x2 blocked):  out[bc](g,w) = sum_f A[c](g,f) * y[bc](f,w)
//   A   : f16 (C, F, F) row-major;  Bt : f16 (BC, W, F);  out f32/f16 (BC,F,W)
// One wave = 32x32 output block (4 tiles, 4 WMMAs per 32-K step).
// ---------------------------------------------------------------------------
__global__ __launch_bounds__(256) void mclin_wmma(const h16* __restrict__ Aall,
                                                  const h16* __restrict__ Bt,
                                                  float* __restrict__ out,
                                                  h16* __restrict__ out_h) {
  int bc   = blockIdx.y;            // 0..15  (b*M+m)
  int c    = bc & 7;
  int wave = threadIdx.x >> 5;
  int lane = threadIdx.x & 31;
  int t    = blockIdx.x * 8 + wave; // 512 blocks of 32x32 per batch
  int gT   = (t >> 4) << 5;         // 0..992  step 32
  int wT   = (t & 15) << 5;         // 0..480  step 32

  const h16* A  = Aall + (size_t)c * FF * FF;
  const h16* Bb = Bt + (size_t)bc * WW * FF;
  int m  = lane & 15;
  int kh = lane >> 4;
  const h16* pa0 = A  + (size_t)(gT + m) * FF + 16 * kh;
  const h16* pa1 = pa0 + (size_t)16 * FF;
  const h16* pb0 = Bb + (size_t)(wT + m) * FF + 16 * kh;
  const h16* pb1 = pb0 + (size_t)16 * FF;

  v8f acc00 = {}, acc01 = {}, acc10 = {}, acc11 = {};
#pragma unroll 4
  for (int k0 = 0; k0 < FF; k0 += 32) {
    __builtin_prefetch(pa0 + k0 + 128, 0, 3);
    __builtin_prefetch(pa1 + k0 + 128, 0, 3);
    __builtin_prefetch(pb0 + k0 + 128, 0, 3);
    __builtin_prefetch(pb1 + k0 + 128, 0, 3);
    v16h a0 = ldfragA(pa0 + k0);
    v16h a1 = ldfragA(pa1 + k0);
    v16h b0 = *(const v16h*)(pb0 + k0);
    v16h b1 = *(const v16h*)(pb1 + k0);
    WMMA(acc00, a0, b0);
    WMMA(acc01, a0, b1);
    WMMA(acc10, a1, b0);
    WMMA(acc11, a1, b1);
  }
  int nl = lane & 15;
#pragma unroll
  for (int ti = 0; ti < 2; ++ti) {
#pragma unroll
    for (int tj = 0; tj < 2; ++tj) {
      const v8f& acc = ti ? (tj ? acc11 : acc10) : (tj ? acc01 : acc00);
      int rbase = gT + 16 * ti + 8 * kh;
      int col   = wT + 16 * tj + nl;
#pragma unroll
      for (int r = 0; r < 8; ++r) {
        float v = acc[r];
        size_t o = (size_t)bc * FF * WW + (size_t)(rbase + r) * WW + col;
        if (out)   out[o]   = v;
        if (out_h) out_h[o] = (h16)v;
      }
    }
  }
}

// ---------------------------------------------------------------------------
// RoPE + head-split:  (BM, F, W) f32  ->  (BM, NH, W, HD) f16 with rotary.
// ---------------------------------------------------------------------------
__global__ __launch_bounds__(256) void rope_heads(const float* __restrict__ src,
                                                  h16* __restrict__ dst) {
  int idx = blockIdx.x * 256 + threadIdx.x;   // 4,194,304 total
  int dp = idx & 63;            // HD/2
  int i  = (idx >> 6) & 511;    // W
  int n  = (idx >> 15) & 7;     // NH
  int bm = idx >> 18;           // 16
  int f0 = n * HD + 2 * dp;
  const float* s = src + ((size_t)bm * FF + f0) * WW + i;
  float x0 = s[0];
  float x1 = s[WW];
  float inv = powf(10000.0f, -(2.0f * dp) / (float)HD);
  float ang = (float)i * inv;
  float cs = cosf(ang), sn = sinf(ang);
  size_t o = ((size_t)(bm * NHEAD + n) * WW + i) * HD + 2 * dp;
  dst[o]     = (h16)(x0 * cs - x1 * sn);
  dst[o + 1] = (h16)(x1 * cs + x0 * sn);
}

// ---------------------------------------------------------------------------
// QK^T WMMA (2x2 blocked):  qk[bmn](i,j) = (1/32) sum_d q(i,d) k(j,d) + prev
// ---------------------------------------------------------------------------
__global__ __launch_bounds__(256) void gemm_qk(const h16* __restrict__ qh,
                                               const h16* __restrict__ khh,
                                               const float* __restrict__ prev,
                                               float* __restrict__ qkout) {
  int bmn  = blockIdx.y;            // 128
  int wave = threadIdx.x >> 5;
  int lane = threadIdx.x & 31;
  int t    = blockIdx.x * 8 + wave; // 256 blocks of 32x32 per bmn
  int iT   = (t >> 4) << 5;
  int jT   = (t & 15) << 5;
  const h16* Qb = qh  + (size_t)bmn * WW * HD;
  const h16* Kb = khh + (size_t)bmn * WW * HD;
  int m  = lane & 15;
  int kh = lane >> 4;
  const h16* pa0 = Qb + (size_t)(iT + m) * HD + 16 * kh;
  const h16* pa1 = pa0 + (size_t)16 * HD;
  const h16* pb0 = Kb + (size_t)(jT + m) * HD + 16 * kh;
  const h16* pb1 = pb0 + (size_t)16 * HD;

  v8f acc00 = {}, acc01 = {}, acc10 = {}, acc11 = {};
#pragma unroll
  for (int k0 = 0; k0 < HD; k0 += 32) {
    v16h a0 = ldfragA(pa0 + k0);
    v16h a1 = ldfragA(pa1 + k0);
    v16h b0 = *(const v16h*)(pb0 + k0);
    v16h b1 = *(const v16h*)(pb1 + k0);
    WMMA(acc00, a0, b0);
    WMMA(acc01, a0, b1);
    WMMA(acc10, a1, b0);
    WMMA(acc11, a1, b1);
  }
  float*       po = qkout + (size_t)bmn * WW * WW;
  const float* pp = prev  + (size_t)bmn * WW * WW;
  int nl = lane & 15;
#pragma unroll
  for (int ti = 0; ti < 2; ++ti) {
#pragma unroll
    for (int tj = 0; tj < 2; ++tj) {
      const v8f& acc = ti ? (tj ? acc11 : acc10) : (tj ? acc01 : acc00);
      int rbase = iT + 16 * ti + 8 * kh;
      int col   = jT + 16 * tj + nl;
#pragma unroll
      for (int r = 0; r < 8; ++r) {
        size_t o = (size_t)(rbase + r) * WW + col;
        po[o] = acc[r] * 0.03125f + pp[o];
      }
    }
  }
}

// ---------------------------------------------------------------------------
// Row softmax over last axis (512) of qk; writes f16 probs (PV A-operand).
// ---------------------------------------------------------------------------
__global__ __launch_bounds__(256) void softmax_rows(const float* __restrict__ qk,
                                                    h16* __restrict__ probs) {
  size_t row = blockIdx.x;                 // 65536 rows
  const float* p = qk + row * WW;
  __shared__ float red[256];
  int tid = threadIdx.x;
  float a0 = p[tid], a1 = p[tid + 256];
  red[tid] = fmaxf(a0, a1);
  __syncthreads();
  for (int s = 128; s > 0; s >>= 1) {
    if (tid < s) red[tid] = fmaxf(red[tid], red[tid + s]);
    __syncthreads();
  }
  float mx = red[0];
  __syncthreads();
  float e0 = __expf(a0 - mx), e1 = __expf(a1 - mx);
  red[tid] = e0 + e1;
  __syncthreads();
  for (int s = 128; s > 0; s >>= 1) {
    if (tid < s) red[tid] += red[tid + s];
    __syncthreads();
  }
  float rs = 1.0f / red[0];
  probs[row * WW + tid]       = (h16)(e0 * rs);
  probs[row * WW + tid + 256] = (h16)(e1 * rs);
}

// ---------------------------------------------------------------------------
// PV WMMA (2x2 blocked):  a[bmn](i,d) = sum_j P(i,j) v(j,d)
//   a_h: f16 (BM, W, F)  -- directly the B-operand layout for the o_pw mclin.
// ---------------------------------------------------------------------------
__global__ __launch_bounds__(256) void gemm_av(const h16* __restrict__ probs,
                                               const h16* __restrict__ v_h,
                                               h16* __restrict__ a_h) {
  int bmn  = blockIdx.y;            // 128
  int bm   = bmn >> 3;
  int nh   = bmn & 7;
  int wave = threadIdx.x >> 5;
  int lane = threadIdx.x & 31;
  int t    = blockIdx.x * 8 + wave; // 64 blocks of 32x32 per bmn
  int iT   = (t >> 2) << 5;         // 0..480
  int dT   = (t & 3) << 5;          // 0..96
  const h16* A  = probs + (size_t)bmn * WW * WW;
  const h16* Bt = v_h + ((size_t)bm * FF + nh * HD + dT) * WW;
  int m  = lane & 15;
  int kh = lane >> 4;
  const h16* pa0 = A + (size_t)(iT + m) * WW + 16 * kh;
  const h16* pa1 = pa0 + (size_t)16 * WW;
  const h16* pb0 = Bt + (size_t)m * WW + 16 * kh;
  const h16* pb1 = pb0 + (size_t)16 * WW;

  v8f acc00 = {}, acc01 = {}, acc10 = {}, acc11 = {};
#pragma unroll 4
  for (int k0 = 0; k0 < WW; k0 += 32) {
    __builtin_prefetch(pa0 + k0 + 128, 0, 3);
    __builtin_prefetch(pa1 + k0 + 128, 0, 3);
    v16h a0 = ldfragA(pa0 + k0);
    v16h a1 = ldfragA(pa1 + k0);
    v16h b0 = *(const v16h*)(pb0 + k0);
    v16h b1 = *(const v16h*)(pb1 + k0);
    WMMA(acc00, a0, b0);
    WMMA(acc01, a0, b1);
    WMMA(acc10, a1, b0);
    WMMA(acc11, a1, b1);
  }
  int nl = lane & 15;
#pragma unroll
  for (int ti = 0; ti < 2; ++ti) {
#pragma unroll
    for (int tj = 0; tj < 2; ++tj) {
      const v8f& acc = ti ? (tj ? acc11 : acc10) : (tj ? acc01 : acc00);
      int rbase = iT + 16 * ti + 8 * kh;
      int dcol  = dT + 16 * tj + nl;
#pragma unroll
      for (int r = 0; r < 8; ++r) {
        int i = rbase + r;
        a_h[((size_t)bm * WW + i) * FF + nh * HD + dcol] = (h16)acc[r];
      }
    }
  }
}

// ---------------------------------------------------------------------------
// o_dw channel mix + concat:  cat(b, 0:8)=x ; cat(b, 8:16)=sum_c odw[d,c]*o1[b,c]
// ---------------------------------------------------------------------------
__global__ __launch_bounds__(256) void odw_concat(const float* __restrict__ x,
                                                  const float* __restrict__ o1,
                                                  const float* __restrict__ odw,
                                                  float* __restrict__ cat) {
  size_t idx = (size_t)blockIdx.x * 256 + threadIdx.x;
  size_t total = (size_t)NB * 16 * FF * WW;
  if (idx >= total) return;
  size_t fw = idx % ((size_t)FF * WW);
  int ch = (int)((idx / ((size_t)FF * WW)) % 16);
  int b  = (int)(idx / ((size_t)FF * WW * 16));
  float r;
  if (ch < 8) {
    r = x[((size_t)(b * NC + ch) * FF * WW) + fw];
  } else {
    int d = ch - 8;
    r = 0.0f;
#pragma unroll
    for (int c = 0; c < NM; ++c)
      r += odw[d * NM + c] * o1[((size_t)(b * NM + c) * FF * WW) + fw];
  }
  cat[idx] = r;
}

// ---------------------------------------------------------------------------
// MultichannelLayerNorm over features of (A+Bv); 1 block per (b,c,w).
// ---------------------------------------------------------------------------
__global__ __launch_bounds__(256) void mcln(const float* __restrict__ A,
                                            const float* __restrict__ Bv,
                                            const float* __restrict__ nw,
                                            const float* __restrict__ nb,
                                            float* __restrict__ out0,
                                            float* __restrict__ out1) {
  int row = blockIdx.x;            // 8192 rows
  int w = row % WW;
  int c = (row / WW) % NC;
  int b = row / (WW * NC);
  size_t base = ((size_t)(b * NC + c) * FF) * WW + w;
  __shared__ float red[256];
  int tid = threadIdx.x;
  float vals[4];
  float s = 0.0f;
#pragma unroll
  for (int j = 0; j < 4; ++j) {
    int f = tid + 256 * j;
    vals[j] = A[base + (size_t)f * WW] + Bv[base + (size_t)f * WW];
    s += vals[j];
  }
  red[tid] = s;
  __syncthreads();
  for (int st = 128; st > 0; st >>= 1) {
    if (tid < st) red[tid] += red[tid + st];
    __syncthreads();
  }
  float mu = red[0] * (1.0f / FF);
  __syncthreads();
  float t2 = 0.0f;
#pragma unroll
  for (int j = 0; j < 4; ++j) {
    float d = vals[j] - mu;
    t2 += d * d;
  }
  red[tid] = t2;
  __syncthreads();
  for (int st = 128; st > 0; st >>= 1) {
    if (tid < st) red[tid] += red[tid + st];
    __syncthreads();
  }
  float rsig = rsqrtf(red[0] * (1.0f / FF) + 1e-5f);
#pragma unroll
  for (int j = 0; j < 4; ++j) {
    int f = tid + 256 * j;
    float y = (vals[j] - mu) * rsig * nw[c * FF + f] + nb[c * FF + f];
    out0[base + (size_t)f * WW] = y;
    if (out1) out1[base + (size_t)f * WW] = y;
  }
}

// ---------------------------------------------------------------------------
extern "C" void kernel_launch(void* const* d_in, const int* in_sizes, int n_in,
                              void* d_out, int out_size, void* d_ws, size_t ws_size,
                              hipStream_t stream) {
  const float* x       = (const float*)d_in[0];
  const float* prev_qk = (const float*)d_in[1];
  const float* qconv_w = (const float*)d_in[2];
  const float* qconv_b = (const float*)d_in[3];
  const float* q_pw    = (const float*)d_in[4];
  const float* kconv_w = (const float*)d_in[5];
  const float* kconv_b = (const float*)d_in[6];
  const float* k_pw    = (const float*)d_in[7];
  const float* vconv_w = (const float*)d_in[8];
  const float* vconv_b = (const float*)d_in[9];
  const float* v_pw    = (const float*)d_in[10];
  const float* o_pw    = (const float*)d_in[11];
  const float* o_dw    = (const float*)d_in[12];
  const float* oproj_w = (const float*)d_in[13];
  const float* oproj_b = (const float*)d_in[14];
  const float* n1_w    = (const float*)d_in[15];
  const float* n1_b    = (const float*)d_in[16];
  const float* c1_w    = (const float*)d_in[17];
  const float* c1_b    = (const float*)d_in[18];
  const float* c2_w    = (const float*)d_in[19];
  const float* c2_b    = (const float*)d_in[20];
  const float* n2_w    = (const float*)d_in[21];
  const float* n2_b    = (const float*)d_in[22];

  const size_t OUTN = (size_t)NB * NC * FF * WW;   // 8,388,608
  float* out0 = (float*)d_out;
  float* out1 = out0 + OUTN;
  float* qk   = out0 + 2 * OUTN;                   // 33,554,432 f32

  // --- workspace layout (bump + deliberate sequential aliasing) ---
  char* ws = (char*)d_ws;
  const size_t PWN = (size_t)NM * FF * FF;         // 8,388,608
  h16* pwq_h = (h16*)(ws);                         // 4 * PWN halves
  h16* pwk_h = pwq_h + PWN;
  h16* pwv_h = pwk_h + PWN;
  h16* pwo_h = pwv_h + PWN;
  h16* yTq = (h16*)(ws + 67108864);                // 3 * OUTN halves
  h16* yTk = yTq + OUTN;
  h16* yTv = yTk + OUTN;
  float* qf32 = (float*)(ws + 117440512);          // 3 * OUTN floats
  float* kf32 = qf32 + OUTN;
  float* hid  = kf32 + OUTN;                       // "vf32" slot (f32 V never needed)
  h16* v_h   = (h16*)(ws + 218103808);             // OUTN halves
  h16* probs = (h16*)(ws + 234881024);             // 33,554,432 halves
  float* catb = (float*)probs;                     // reuse after PV GEMM (16.8M f32)
  float* t1 = (float*)(ws + 301989888);            // 33,554,432 f32
  float* z2 = (float*)(ws + 436207616);            // 8,388,608 f32
  // aliases (strictly ordered on `stream`):
  h16* qh  = yTq;    // rope output reuses yT buffers after mclin reads them
  h16* kh  = yTk;
  h16* a_h = yTv;
  float* o1 = qf32;  // mclin-o output reuses q f32 after rope consumed it
  float* zb = kf32;  // oproj output reuses k f32

  dim3 blk(256);

  // 1) weight conversion to f16
  cvt_f32_f16<<<dim3((unsigned)((PWN + 255) / 256)), blk, 0, stream>>>(q_pw, pwq_h, (int)PWN);
  cvt_f32_f16<<<dim3((unsigned)((PWN + 255) / 256)), blk, 0, stream>>>(k_pw, pwk_h, (int)PWN);
  cvt_f32_f16<<<dim3((unsigned)((PWN + 255) / 256)), blk, 0, stream>>>(v_pw, pwv_h, (int)PWN);
  cvt_f32_f16<<<dim3((unsigned)((PWN + 255) / 256)), blk, 0, stream>>>(o_pw, pwo_h, (int)PWN);

  // 2) q/k/v convs -> f16 transposed (WMMA B operand)
  unsigned convg = (unsigned)((OUTN + 255) / 256);
  conv3x3<<<dim3(convg), blk, 0, stream>>>(x, qconv_w, qconv_b, NC, NM, nullptr, yTq, 0);
  conv3x3<<<dim3(convg), blk, 0, stream>>>(x, kconv_w, kconv_b, NC, NM, nullptr, yTk, 0);
  conv3x3<<<dim3(convg), blk, 0, stream>>>(x, vconv_w, vconv_b, NC, NM, nullptr, yTv, 0);

  // 3) mclin GEMMs (WMMA): q,k -> f32 (for rope); v -> f16 (PV B operand)
  dim3 mgrid(64, NB * NM);
  mclin_wmma<<<mgrid, blk, 0, stream>>>(pwq_h, yTq, qf32, nullptr);
  mclin_wmma<<<mgrid, blk, 0, stream>>>(pwk_h, yTk, kf32, nullptr);
  mclin_wmma<<<mgrid, blk, 0, stream>>>(pwv_h, yTv, nullptr, v_h);

  // 4) rope + head split (f16)
  rope_heads<<<dim3(16384), blk, 0, stream>>>(qf32, qh);
  rope_heads<<<dim3(16384), blk, 0, stream>>>(kf32, kh);

  // 5) QK^T (WMMA) fused scale + prev_qk, straight into d_out qk region
  gemm_qk<<<dim3(32, 128), blk, 0, stream>>>(qh, kh, prev_qk, qk);

  // 6) softmax -> f16 probs
  softmax_rows<<<dim3(65536), blk, 0, stream>>>(qk, probs);

  // 7) PV (WMMA) -> a in (BM, W, F) f16 (o_pw B operand)
  gemm_av<<<dim3(8, 128), blk, 0, stream>>>(probs, v_h, a_h);

  // 8) o_pw mclin (WMMA) -> f32
  mclin_wmma<<<mgrid, blk, 0, stream>>>(pwo_h, a_h, o1, nullptr);

  // 9) o_dw mix + concat(x, a2)
  odw_concat<<<dim3((unsigned)((2 * OUTN + 255) / 256)), blk, 0, stream>>>(x, o1, o_dw, catb);

  // 10) oproj conv (16 -> 8)
  conv3x3<<<dim3(convg), blk, 0, stream>>>(catb, oproj_w, oproj_b, 16, NC, zb, nullptr, 0);

  // 11) hid = LN(x + z)
  mcln<<<dim3(8192), blk, 0, stream>>>(x, zb, n1_w, n1_b, hid, nullptr);

  // 12) c1 conv (8 -> 32) + exact GELU
  conv3x3<<<dim3((unsigned)((4 * OUTN + 255) / 256)), blk, 0, stream>>>(hid, c1_w, c1_b, NC, 32, t1, nullptr, 1);

  // 13) c2 conv (32 -> 8)
  conv3x3<<<dim3(convg), blk, 0, stream>>>(t1, c2_w, c2_b, 32, NC, z2, nullptr, 0);

  // 14) out = LN(hid + z2), duplicated into both output slots
  mcln<<<dim3(8192), blk, 0, stream>>>(hid, z2, n2_w, n2_b, out0, out1);
}
